// attention_80513456931144
// MI455X (gfx1250) — compile-verified, matrix operand order checked by reading, not run on hardware
//
#include <hip/hip_runtime.h>
#include <hip/hip_bf16.h>

typedef __attribute__((ext_vector_type(16))) __bf16 v16bf;
typedef __attribute__((ext_vector_type(8)))  __bf16 bf16x8;
typedef __attribute__((ext_vector_type(4)))  __bf16 bf16x4;
typedef __attribute__((ext_vector_type(8)))  float  v8f;
typedef __attribute__((ext_vector_type(4)))  float  f32x4;

#define EMBED 1024
#define HEADS 16
#define HDIM  64
#define NBATCH 4
#define SEQ   2048
#define MTOT  (NBATCH * SEQ)   // 8192

// ---------------------------------------------------------------------------
// WMMA helpers (CDNA5 wave32, V_WMMA_F32_16X16X32_BF16)
// A fragment (16x32 bf16): lane L (L&15 = row M), g = L>>4; half j holds
// K = (j&7) + 16*(j>>3) + 8g -> two 16B chunks at K offsets 8g and 16+8g.
// B fragment (32x16) is identical with lane&15 = column N (rows of B^T).
// ---------------------------------------------------------------------------
__device__ __forceinline__ v8f wmma_bf16(v16bf a, v16bf b, v8f c) {
  return __builtin_amdgcn_wmma_f32_16x16x32_bf16(
      false, a, false, b, (short)0, c, false, false);
}

__device__ __forceinline__ v16bf load_frag(const __bf16* p, int ld) {
  const int lane = threadIdx.x & 31;
  const __bf16* q = p + (lane & 15) * ld + ((lane >> 4) << 3);
  bf16x8 lo = *reinterpret_cast<const bf16x8*>(q);
  bf16x8 hi = *reinterpret_cast<const bf16x8*>(q + 16);
  v16bf r;
#pragma unroll
  for (int i = 0; i < 8; ++i) { r[i] = lo[i]; r[i + 8] = hi[i]; }
  return r;
}

// ---------------------------------------------------------------------------
// xor-butterfly within each 16-lane half. gfx12-family has v_permlane16_b32
// (VALU): same 16-nibble selector applied to lanes 0-15 and 16-31 -> no DS
// traffic, no s_wait_dscnt serialization (unlike __shfl_xor -> ds_bpermute).
// ---------------------------------------------------------------------------
#if __has_builtin(__builtin_amdgcn_permlane16)
template <int OFF> __device__ __forceinline__ float xor16(float x) {
  unsigned u = __float_as_uint(x);
  unsigned s0, s1;  // nibble i = source lane id (i^OFF) within the 16-group
  if constexpr (OFF == 8)      { s0 = 0xFEDCBA98u; s1 = 0x76543210u; }
  else if constexpr (OFF == 4) { s0 = 0x32107654u; s1 = 0xBA98FEDCu; }
  else if constexpr (OFF == 2) { s0 = 0x54761032u; s1 = 0xDCFE98BAu; }
  else                         { s0 = 0x67452301u; s1 = 0xEFCDAB89u; }
  return __uint_as_float(__builtin_amdgcn_permlane16(u, u, s0, s1, false, false));
}
#else
template <int OFF> __device__ __forceinline__ float xor16(float x) {
  return __shfl_xor(x, OFF, 16);
}
#endif

// ---------------------------------------------------------------------------
// GEMM: C[M=8192][N=1024] = A[M][1024] @ W^T, W stored [out][in] row-major.
// Block tile 128x128, K-step 64, 256 thr = 8 waves, wave tile 32x64.
// Double-buffered LDS: one barrier per K-step, global loads overlapped.
// MODE 0: f32 query -> Q bf16 [n][h][s][d] scaled 1/32
// MODE 1: f32 keys  -> K bf16 [n][h][s][d]
// MODE 2: f32 values-> V^T bf16 [n][h][d][s]
// MODE 3: bf16 attn-out [m][e] -> f32 d_out [m][e]
// ---------------------------------------------------------------------------
template <int MODE>
__global__ __launch_bounds__(256) void gemm128(const float* __restrict__ A32,
                                               const __bf16* __restrict__ A16,
                                               const float* __restrict__ W,
                                               float* __restrict__ outf,
                                               __bf16* __restrict__ outbf) {
  constexpr int LDT = 72;          // 64 + 8 pad; 144B rows keep 16B alignment
  constexpr int BK  = 64;
  constexpr int NK  = EMBED / BK;  // 16
  __shared__ __bf16 As[2][128 * LDT];
  __shared__ __bf16 Bs[2][128 * LDT];

  const int tid = threadIdx.x;
  const int w = tid >> 5, wr = w & 3, wc = w >> 2;
  const int m0 = blockIdx.y * 128, n0 = blockIdx.x * 128;

  const v8f vzero = {0.f, 0.f, 0.f, 0.f, 0.f, 0.f, 0.f, 0.f};
  v8f acc[2][4];
#pragma unroll
  for (int mi = 0; mi < 2; ++mi)
#pragma unroll
    for (int ni = 0; ni < 4; ++ni) acc[mi][ni] = vzero;

  bf16x4 aR[8], bR[8];   // f32->bf16 staged tiles (32 elem/thread each)
  bf16x8 aR3[4];         // MODE 3: bf16 A staging

  auto gloadA = [&](int k0) {
    if constexpr (MODE == 3) {
#pragma unroll
      for (int i = 0; i < 4; ++i) {
        int c = tid + i * 256, row = c >> 3, col = (c & 7) << 3;
        aR3[i] = *reinterpret_cast<const bf16x8*>(
            &A16[(size_t)(m0 + row) * EMBED + k0 + col]);
      }
    } else {
#pragma unroll
      for (int i = 0; i < 8; ++i) {
        int c = tid + i * 256, row = c >> 4, col = (c & 15) << 2;
        f32x4 v = *reinterpret_cast<const f32x4*>(
            &A32[(size_t)(m0 + row) * EMBED + k0 + col]);
        bf16x4 b;
#pragma unroll
        for (int j = 0; j < 4; ++j) b[j] = (__bf16)v[j];
        aR[i] = b;
      }
    }
  };
  auto gloadB = [&](int k0) {
#pragma unroll
    for (int i = 0; i < 8; ++i) {
      int c = tid + i * 256, row = c >> 4, col = (c & 15) << 2;
      f32x4 v = *reinterpret_cast<const f32x4*>(
          &W[(size_t)(n0 + row) * EMBED + k0 + col]);
      bf16x4 b;
#pragma unroll
      for (int j = 0; j < 4; ++j) b[j] = (__bf16)v[j];
      bR[i] = b;
    }
  };
  auto storeLds = [&](int buf) {
    if constexpr (MODE == 3) {
#pragma unroll
      for (int i = 0; i < 4; ++i) {
        int c = tid + i * 256, row = c >> 3, col = (c & 7) << 3;
        *reinterpret_cast<bf16x8*>(&As[buf][row * LDT + col]) = aR3[i];
      }
    } else {
#pragma unroll
      for (int i = 0; i < 8; ++i) {
        int c = tid + i * 256, row = c >> 4, col = (c & 15) << 2;
        *reinterpret_cast<bf16x4*>(&As[buf][row * LDT + col]) = aR[i];
      }
    }
#pragma unroll
    for (int i = 0; i < 8; ++i) {
      int c = tid + i * 256, row = c >> 4, col = (c & 15) << 2;
      *reinterpret_cast<bf16x4*>(&Bs[buf][row * LDT + col]) = bR[i];
    }
  };

  gloadA(0);
  gloadB(0);
  storeLds(0);
  __syncthreads();

  for (int kb = 0; kb < NK; ++kb) {
    const int cur = kb & 1;
    if (kb + 1 < NK) { gloadA((kb + 1) * BK); gloadB((kb + 1) * BK); }
    if (kb + 2 < NK) {  // L2 warm-up -> global_prefetch_b8
      int row = tid >> 4, col = (tid & 15) << 2;
      __builtin_prefetch(&W[(size_t)(n0 + row) * EMBED + (kb + 2) * BK + col], 0, 1);
      if constexpr (MODE == 3)
        __builtin_prefetch(&A16[(size_t)(m0 + (tid >> 3)) * EMBED + (kb + 2) * BK], 0, 1);
      else
        __builtin_prefetch(&A32[(size_t)(m0 + row) * EMBED + (kb + 2) * BK + col], 0, 1);
    }
#pragma unroll
    for (int kk = 0; kk < 2; ++kk) {
      v16bf af[2];
#pragma unroll
      for (int mi = 0; mi < 2; ++mi)
        af[mi] = load_frag(&As[cur][(wr * 32 + mi * 16) * LDT + kk * 32], LDT);
#pragma unroll
      for (int ni = 0; ni < 4; ++ni) {
        v16bf bfr = load_frag(&Bs[cur][(wc * 64 + ni * 16) * LDT + kk * 32], LDT);
#pragma unroll
        for (int mi = 0; mi < 2; ++mi)
          acc[mi][ni] = wmma_bf16(af[mi], bfr, acc[mi][ni]);
      }
    }
    if (kb + 1 < NK) storeLds(cur ^ 1);
    __syncthreads();
  }

  // epilogue: C layout = VGPR r, lanes0-15 (M=r), lanes16-31 (M=r+8)
  const int lane = tid & 31;
  const int cn = lane & 15, cg = lane >> 4;
#pragma unroll
  for (int mi = 0; mi < 2; ++mi)
#pragma unroll
    for (int ni = 0; ni < 4; ++ni)
#pragma unroll
      for (int r = 0; r < 8; ++r) {
        int m = m0 + wr * 32 + mi * 16 + r + cg * 8;
        int n = n0 + wc * 64 + ni * 16 + cn;
        float v = acc[mi][ni][r];
        if constexpr (MODE == 0) v *= 0.03125f;  // 1/sqrt(EMBED)
        if constexpr (MODE == 0 || MODE == 1) {
          int nb = m >> 11, s = m & 2047, h = n >> 6, d = n & 63;
          outbf[((size_t)((nb * HEADS + h) * SEQ + s)) * HDIM + d] = (__bf16)v;
        } else if constexpr (MODE == 2) {
          int nb = m >> 11, s = m & 2047, h = n >> 6, d = n & 63;
          outbf[((size_t)((nb * HEADS + h) * HDIM + d)) * SEQ + s] = (__bf16)v;
        } else {
          outf[(size_t)m * EMBED + n] = v;
        }
      }
}

// ---------------------------------------------------------------------------
// Flash attention: block = 128 queries of one (batch, head); 8 waves, each
// owns 16 query rows. K/V double-buffered in LDS (one barrier per 64-key
// block); online softmax in registers with permlane16 reductions; P goes
// C-layout -> A-layout through per-wave LDS. Writes bf16 AO [n][s][h*D+d].
// ---------------------------------------------------------------------------
__global__ __launch_bounds__(256) void flash_attn(const __bf16* __restrict__ Q,
                                                  const __bf16* __restrict__ K,
                                                  const __bf16* __restrict__ Vt,
                                                  __bf16* __restrict__ AO) {
  constexpr int LDT = 72;
  __shared__ __bf16 Ks[2][64 * LDT];
  __shared__ __bf16 Vs[2][64 * LDT];
  __shared__ __bf16 Ps[8 * 16 * LDT];

  const int tid = threadIdx.x;
  const int w = tid >> 5;
  const int lane = tid & 31;
  const int cn = lane & 15, cg = lane >> 4;

  const int b    = blockIdx.x;
  const int qblk = b & 15;
  const int h    = (b >> 4) & 15;
  const int nb   = b >> 8;

  const __bf16* Qh = Q  + (size_t)((nb * HEADS + h) * SEQ) * HDIM;
  const __bf16* Kh = K  + (size_t)((nb * HEADS + h) * SEQ) * HDIM;
  const __bf16* Vh = Vt + (size_t)((nb * HEADS + h) * HDIM) * SEQ;
  const int q0 = qblk * 128 + w * 16;

  v16bf qf[2];
#pragma unroll
  for (int kk = 0; kk < 2; ++kk)
    qf[kk] = load_frag(Qh + (size_t)q0 * HDIM + kk * 32, HDIM);

  const v8f vzero = {0.f, 0.f, 0.f, 0.f, 0.f, 0.f, 0.f, 0.f};
  v8f o[4];
#pragma unroll
  for (int t = 0; t < 4; ++t) o[t] = vzero;
  float Mrow[8], Lrow[8];
#pragma unroll
  for (int r = 0; r < 8; ++r) { Mrow[r] = -__builtin_inff(); Lrow[r] = 0.f; }

  __bf16* Pw = &Ps[w * 16 * LDT];

  bf16x8 kreg[2], vreg[2];
  auto gload = [&](int s0) {
#pragma unroll
    for (int i = 0; i < 2; ++i) {
      int c = tid + i * 256, row = c >> 3, col = (c & 7) << 3;
      kreg[i] = *reinterpret_cast<const bf16x8*>(&Kh[(size_t)(s0 + row) * HDIM + col]);
      vreg[i] = *reinterpret_cast<const bf16x8*>(&Vh[(size_t)row * SEQ + s0 + col]);
    }
  };
  auto stage = [&](int buf) {
#pragma unroll
    for (int i = 0; i < 2; ++i) {
      int c = tid + i * 256, row = c >> 3, col = (c & 7) << 3;
      *reinterpret_cast<bf16x8*>(&Ks[buf][row * LDT + col]) = kreg[i];
      *reinterpret_cast<bf16x8*>(&Vs[buf][row * LDT + col]) = vreg[i];
    }
  };

  gload(0);
  stage(0);
  __syncthreads();

  constexpr int NBLK = SEQ / 64;   // 32
  for (int kb = 0; kb < NBLK; ++kb) {
    const int cur = kb & 1;
    if (kb + 1 < NBLK) gload((kb + 1) * 64);

    // energy tile 16 x 64 (Q already carries 1/sqrt(E))
    v8f e[4];
#pragma unroll
    for (int t = 0; t < 4; ++t) e[t] = vzero;
#pragma unroll
    for (int kk = 0; kk < 2; ++kk)
#pragma unroll
      for (int nt = 0; nt < 4; ++nt) {
        v16bf bfrag = load_frag(&Ks[cur][(nt * 16) * LDT + kk * 32], LDT);
        e[nt] = wmma_bf16(qf[kk], bfrag, e[nt]);
      }

    // ---- online softmax, level-major for ILP; VALU permlane reductions ----
    float mx[8], sc[8], rs[8];
#pragma unroll
    for (int r = 0; r < 8; ++r)
      mx[r] = fmaxf(fmaxf(e[0][r], e[1][r]), fmaxf(e[2][r], e[3][r]));
#pragma unroll
    for (int r = 0; r < 8; ++r) mx[r] = fmaxf(mx[r], xor16<8>(mx[r]));
#pragma unroll
    for (int r = 0; r < 8; ++r) mx[r] = fmaxf(mx[r], xor16<4>(mx[r]));
#pragma unroll
    for (int r = 0; r < 8; ++r) mx[r] = fmaxf(mx[r], xor16<2>(mx[r]));
#pragma unroll
    for (int r = 0; r < 8; ++r) mx[r] = fmaxf(mx[r], xor16<1>(mx[r]));
#pragma unroll
    for (int r = 0; r < 8; ++r) {
      float newM = fmaxf(Mrow[r], mx[r]);
      sc[r] = __expf(Mrow[r] - newM);
      Mrow[r] = newM;
    }
#pragma unroll
    for (int r = 0; r < 8; ++r) {
      float s = 0.f;
#pragma unroll
      for (int t = 0; t < 4; ++t) {
        float p = __expf(e[t][r] - Mrow[r]);
        e[t][r] = p;
        s += p;
      }
      rs[r] = s;
    }
#pragma unroll
    for (int r = 0; r < 8; ++r) rs[r] += xor16<8>(rs[r]);
#pragma unroll
    for (int r = 0; r < 8; ++r) rs[r] += xor16<4>(rs[r]);
#pragma unroll
    for (int r = 0; r < 8; ++r) rs[r] += xor16<2>(rs[r]);
#pragma unroll
    for (int r = 0; r < 8; ++r) rs[r] += xor16<1>(rs[r]);
#pragma unroll
    for (int r = 0; r < 8; ++r) Lrow[r] = Lrow[r] * sc[r] + rs[r];

    // rescale O and spill P to LDS (C-layout -> A-layout transpose)
#pragma unroll
    for (int t = 0; t < 4; ++t)
#pragma unroll
      for (int r = 0; r < 8; ++r) {
        o[t][r] *= sc[r];
        Pw[(r + 8 * cg) * LDT + t * 16 + cn] = (__bf16)e[t][r];
      }
    asm volatile("s_wait_dscnt 0" ::: "memory");  // same-wave LDS RAW

    // O += P(16x64) @ V(64x64)
#pragma unroll
    for (int kk = 0; kk < 2; ++kk) {
      v16bf pf = load_frag(Pw + kk * 32, LDT);
#pragma unroll
      for (int dt = 0; dt < 4; ++dt) {
        v16bf vf = load_frag(&Vs[cur][(dt * 16) * LDT + kk * 32], LDT);
        o[dt] = wmma_bf16(pf, vf, o[dt]);
      }
    }

    if (kb + 1 < NBLK) stage(cur ^ 1);
    __syncthreads();
  }

  // epilogue: normalize, write bf16 AO[n][s][h*D + d]
#pragma unroll
  for (int t = 0; t < 4; ++t)
#pragma unroll
    for (int r = 0; r < 8; ++r) {
      float v = o[t][r] / Lrow[r];
      int s = q0 + r + 8 * cg;
      int ecol = h * HDIM + t * 16 + cn;
      AO[((size_t)nb * SEQ + s) * EMBED + ecol] = (__bf16)v;
    }
}

// ---------------------------------------------------------------------------
extern "C" void kernel_launch(void* const* d_in, const int* in_sizes, int n_in,
                              void* d_out, int out_size, void* d_ws, size_t ws_size,
                              hipStream_t stream) {
  (void)in_sizes; (void)n_in; (void)out_size; (void)ws_size;
  const float* values_in = (const float*)d_in[0];
  const float* keys_in   = (const float*)d_in[1];
  const float* query_in  = (const float*)d_in[2];
  const float* W_v = (const float*)d_in[3];
  const float* W_k = (const float*)d_in[4];
  const float* W_q = (const float*)d_in[5];
  const float* W_o = (const float*)d_in[6];

  const size_t NHSD = (size_t)NBATCH * HEADS * SEQ * HDIM;  // 8,388,608
  __bf16* Qb  = (__bf16*)d_ws;
  __bf16* Kb  = Qb + NHSD;
  __bf16* Vtb = Kb + NHSD;
  __bf16* AOb = Vtb + NHSD;   // 64 MiB scratch total

  dim3 grid(EMBED / 128, MTOT / 128);  // (8, 64)
  gemm128<0><<<grid, 256, 0, stream>>>(query_in,  nullptr, W_q, nullptr, Qb);
  gemm128<1><<<grid, 256, 0, stream>>>(keys_in,   nullptr, W_k, nullptr, Kb);
  gemm128<2><<<grid, 256, 0, stream>>>(values_in, nullptr, W_v, nullptr, Vtb);

  flash_attn<<<NBATCH * HEADS * (SEQ / 128), 256, 0, stream>>>(Qb, Kb, Vtb, AOb);

  gemm128<3><<<grid, 256, 0, stream>>>(nullptr, AOb, W_o, (float*)d_out, nullptr);
}
